// Encoder_58566174048378
// MI455X (gfx1250) — matrix operand order repaired and checked by reading, not compiled, not run on hardware
//
#include <hip/hip_runtime.h>
#include <hip/hip_bf16.h>
#include <stdint.h>

typedef __bf16 bf16;
typedef __bf16 v16bf __attribute__((ext_vector_type(16)));
typedef float  v8f   __attribute__((ext_vector_type(8)));

static constexpr int T   = 128;
static constexpr int BSZ = 256;
static constexpr int IN  = 512;
static constexpr int H1  = 1024;
static constexpr int H2  = 1024;
static constexpr int H3  = 512;

// CDNA5 async global->LDS copy, 16 bytes per lane, tracked by ASYNCcnt.
// saddr form: VDST = LDS byte-address VGPR, VADDR = 32-bit byte-offset VGPR,
// SADDR = uniform 64-bit base (SGPR pair).
__device__ __forceinline__ void async_copy_b128(unsigned lds_addr, unsigned voff,
                                                const void* sbase)
{
    asm volatile("global_load_async_to_lds_b128 %0, %1, %2"
                 :: "v"(lds_addr), "v"(voff), "s"(sbase)
                 : "memory");
}

__device__ __forceinline__ void wait_async_zero()
{
    asm volatile("s_wait_asynccnt 0x0" ::: "memory");
}

// ---------------------------------------------------------------------------
// GEMM: gates[m][n] = sum_k A[m][k] * W[n][k]   (A = [A0 | A1] along k)
// M = 256 fixed. Tile: BM=128, BN=64. K staged 64 at a time into
// double-buffered LDS via async-to-LDS DMA; consumed as 2 x (k=32) WMMA steps.
// 256 threads = 8 waves; wave w computes rows [16w,16w+16) x 64 cols
// = 4 accumulators of v_wmma_f32_16x16x32_bf16 (8 WMMAs per barrier pair).
// ---------------------------------------------------------------------------
__global__ __launch_bounds__(256)
void lstm_gates_gemm(const bf16* __restrict__ A0, int lda0, int Ksplit,
                     const bf16* __restrict__ A1, int lda1,
                     const bf16* __restrict__ W,  int K, int N,
                     float* __restrict__ Cg)
{
    __shared__ __align__(16) bf16 As[2][128][64];   // 2 x 16 KB
    __shared__ __align__(16) bf16 Bs[2][64][64];    // 2 x  8 KB

    const int tid    = threadIdx.x;
    const int nblk   = blockIdx.x * 64;
    const int mblk   = blockIdx.y * 128;
    const int wave   = tid >> 5;
    const int lane   = tid & 31;
    const int lane16 = lane & 15;
    const int hi     = lane >> 4;        // 0: lanes 0-15, 1: lanes 16-31

    // Issue one 64-wide K stage (A: 128x64, B: 64x64) into LDS buffer `buf`.
    // Ksplit and K are multiples of 64, so a stage never straddles A0/A1.
    auto issue_stage = [&](int buf, int k0) {
        const bf16* srcA;
        int lda, kcol;
        if (k0 < Ksplit) { srcA = A0; lda = lda0; kcol = k0; }
        else             { srcA = A1; lda = lda1; kcol = k0 - Ksplit; }
#pragma unroll
        for (int i = 0; i < 4; ++i) {                 // 1024 chunks of 16B
            int c   = tid + i * 256;
            int row = c >> 3;
            int cp  = c & 7;
            unsigned voff = (unsigned)(((size_t)(mblk + row) * lda + kcol + cp * 8)
                                       * sizeof(bf16));
            unsigned lds  = (unsigned)(uintptr_t)(&As[buf][row][cp * 8]);
            async_copy_b128(lds, voff, srcA);
        }
#pragma unroll
        for (int i = 0; i < 2; ++i) {                 // 512 chunks of 16B
            int c   = tid + i * 256;
            int row = c >> 3;
            int cp  = c & 7;
            unsigned voff = (unsigned)(((size_t)(nblk + row) * K + k0 + cp * 8)
                                       * sizeof(bf16));
            unsigned lds  = (unsigned)(uintptr_t)(&Bs[buf][row][cp * 8]);
            async_copy_b128(lds, voff, W);
        }
    };

    v8f acc0 = {}, acc1 = {}, acc2 = {}, acc3 = {};

    issue_stage(0, 0);                 // prologue prefetch
    int buf = 0;

    for (int k0 = 0; k0 < K; k0 += 64) {
        wait_async_zero();             // current buffer's DMA complete
        __syncthreads();               // visible to all waves
        if (k0 + 64 < K) issue_stage(buf ^ 1, k0 + 64);   // prefetch next stage

        const uint4* pa  = (const uint4*)(&As[buf][wave * 16 + lane16][0]);
        const uint4* pb0 = (const uint4*)(&Bs[buf][ 0 + lane16][0]);
        const uint4* pb1 = (const uint4*)(&Bs[buf][16 + lane16][0]);
        const uint4* pb2 = (const uint4*)(&Bs[buf][32 + lane16][0]);
        const uint4* pb3 = (const uint4*)(&Bs[buf][48 + lane16][0]);

#pragma unroll
        for (int s = 0; s < 2; ++s) {  // two k=32 sub-steps per stage
            // A fragment (16x32): lanes0-15 K 0..15,
            // lanes16-31 K 16..31 -> two ds_read_b128
            union { uint4 u[2]; v16bf v; } fa, fb0, fb1, fb2, fb3;
            fa.u[0] = pa[s * 4 + 2 * hi];
            fa.u[1] = pa[s * 4 + 2 * hi + 1];
            // B fragments (32x16): lanes0-15 hold K 0..15, lanes16-31 K 16..31
            fb0.u[0] = pb0[s * 4 + 2 * hi]; fb0.u[1] = pb0[s * 4 + 2 * hi + 1];
            fb1.u[0] = pb1[s * 4 + 2 * hi]; fb1.u[1] = pb1[s * 4 + 2 * hi + 1];
            fb2.u[0] = pb2[s * 4 + 2 * hi]; fb2.u[1] = pb2[s * 4 + 2 * hi + 1];
            fb3.u[0] = pb3[s * 4 + 2 * hi]; fb3.u[1] = pb3[s * 4 + 2 * hi + 1];

            acc0 = __builtin_amdgcn_wmma_f32_16x16x32_bf16(false, fa.v, false, fb0.v,
                                                           (short)0, acc0, false, false);
            acc1 = __builtin_amdgcn_wmma_f32_16x16x32_bf16(false, fa.v, false, fb1.v,
                                                           (short)0, acc1, false, false);
            acc2 = __builtin_amdgcn_wmma_f32_16x16x32_bf16(false, fa.v, false, fb2.v,
                                                           (short)0, acc2, false, false);
            acc3 = __builtin_amdgcn_wmma_f32_16x16x32_bf16(false, fa.v, false, fb3.v,
                                                           (short)0, acc3, false, false);
        }
        __syncthreads();               // all reads of `buf` done before reuse
        buf ^= 1;
    }

    // ---- store C: lane n = lane16, VGPR r -> m = r + 8*hi ----
    const int ncol = nblk + lane16;
#pragma unroll
    for (int r = 0; r < 8; ++r) {
        const size_t m = (size_t)(mblk + wave * 16 + r + hi * 8);
        Cg[m * N + ncol +  0] = acc0[r];
        Cg[m * N + ncol + 16] = acc1[r];
        Cg[m * N + ncol + 32] = acc2[r];
        Cg[m * N + ncol + 48] = acc3[r];
    }
}

// ---------------------------------------------------------------------------
// Pointwise LSTM cell update (gate order i,f,g,o).  c,h are fp32 state
// (h,c live in d_out); hbf is the bf16 copy feeding the next GEMMs.
// ---------------------------------------------------------------------------
__global__ __launch_bounds__(256)
void lstm_pointwise(const float* __restrict__ gates, const float* __restrict__ bias,
                    float* __restrict__ c, float* __restrict__ h,
                    bf16* __restrict__ hbf, int H)
{
    int idx = blockIdx.x * blockDim.x + threadIdx.x;
    if (idx >= BSZ * H) return;
    int b = idx / H, n = idx % H;
    const float* gr = gates + (size_t)b * 4 * H;
    float gi = gr[n]         + bias[n];
    float gf = gr[H + n]     + bias[H + n];
    float gg = gr[2 * H + n] + bias[2 * H + n];
    float go = gr[3 * H + n] + bias[3 * H + n];
    float si = 1.0f / (1.0f + __expf(-gi));
    float sf = 1.0f / (1.0f + __expf(-gf));
    float so = 1.0f / (1.0f + __expf(-go));
    float cn = sf * c[idx] + si * tanhf(gg);
    float hn = so * tanhf(cn);
    c[idx]   = cn;
    h[idx]   = hn;
    hbf[idx] = (bf16)hn;
}

// ---------------------------------------------------------------------------
// Prep kernels
// ---------------------------------------------------------------------------
__global__ void cvt_bf16_k(const float* __restrict__ in, bf16* __restrict__ out, long n)
{
    long i = (long)blockIdx.x * blockDim.x + threadIdx.x;
    long stride = (long)gridDim.x * blockDim.x;
    for (; i < n; i += stride) out[i] = (bf16)in[i];
}

// Wcat[n][k] = k < in_dim ? Wih[n][k] : Whh[n][k-in_dim]   (bf16, row-major K)
__global__ void build_wcat_k(const float* __restrict__ Wih, const float* __restrict__ Whh,
                             bf16* __restrict__ Wcat, int in_dim, int Hd, int K, long total)
{
    long i = (long)blockIdx.x * blockDim.x + threadIdx.x;
    long stride = (long)gridDim.x * blockDim.x;
    for (; i < total; i += stride) {
        long n = i / K;
        int  k = (int)(i % K);
        float v = (k < in_dim) ? Wih[n * in_dim + k] : Whh[n * Hd + (k - in_dim)];
        Wcat[i] = (bf16)v;
    }
}

__global__ void bias_sum_k(const float* a, const float* b, float* o, int n)
{
    int i = blockIdx.x * blockDim.x + threadIdx.x;
    if (i < n) o[i] = a[i] + b[i];
}

__global__ void zero_f32_k(float* p, long n)
{
    long i = (long)blockIdx.x * blockDim.x + threadIdx.x;
    long stride = (long)gridDim.x * blockDim.x;
    for (; i < n; i += stride) p[i] = 0.0f;
}

__global__ void zero_bf16_k(bf16* p, long n)
{
    long i = (long)blockIdx.x * blockDim.x + threadIdx.x;
    long stride = (long)gridDim.x * blockDim.x;
    for (; i < n; i += stride) p[i] = (bf16)0.0f;
}

// ---------------------------------------------------------------------------
extern "C" void kernel_launch(void* const* d_in, const int* in_sizes, int n_in,
                              void* d_out, int out_size, void* d_ws, size_t ws_size,
                              hipStream_t stream)
{
    const float* x    = (const float*)d_in[0];
    const float* Wih1 = (const float*)d_in[1];
    const float* Whh1 = (const float*)d_in[2];
    const float* bih1 = (const float*)d_in[3];
    const float* bhh1 = (const float*)d_in[4];
    const float* Wih2 = (const float*)d_in[5];
    const float* Whh2 = (const float*)d_in[6];
    const float* bih2 = (const float*)d_in[7];
    const float* bhh2 = (const float*)d_in[8];
    const float* Wih3 = (const float*)d_in[9];
    const float* Whh3 = (const float*)d_in[10];
    const float* bih3 = (const float*)d_in[11];
    const float* bhh3 = (const float*)d_in[12];

    // ---- workspace layout (bytes) ----
    uint8_t* ws = (uint8_t*)d_ws;
    size_t off = 0;
    auto alloc = [&](size_t bytes) { uint8_t* p = ws + off; off = (off + bytes + 255) & ~(size_t)255; return p; };
    bf16*  xbf  = (bf16*) alloc((size_t)T * BSZ * IN * 2);       // 33.6 MB
    bf16*  w1   = (bf16*) alloc((size_t)4 * H1 * (IN + H1) * 2); // 12.6 MB
    bf16*  w2   = (bf16*) alloc((size_t)4 * H2 * (H1 + H2) * 2); // 16.8 MB
    bf16*  w3   = (bf16*) alloc((size_t)4 * H3 * (H2 + H3) * 2); //  6.3 MB
    float* b1   = (float*)alloc((size_t)4 * H1 * 4);
    float* b2   = (float*)alloc((size_t)4 * H2 * 4);
    float* b3   = (float*)alloc((size_t)4 * H3 * 4);
    float* gbuf = (float*)alloc((size_t)BSZ * 4 * H1 * 4);       //  4.2 MB
    bf16*  h1bf = (bf16*) alloc((size_t)BSZ * H1 * 2);
    bf16*  h2bf = (bf16*) alloc((size_t)BSZ * H2 * 2);
    bf16*  h3bf = (bf16*) alloc((size_t)BSZ * H3 * 2);
    (void)ws_size; (void)n_in; (void)in_sizes;

    // ---- fp32 state lives directly in d_out: h1,c1,h2,c2,h3,c3 ----
    float* h1f = (float*)d_out;
    float* c1f = h1f + (size_t)BSZ * H1;
    float* h2f = c1f + (size_t)BSZ * H1;
    float* c2f = h2f + (size_t)BSZ * H2;
    float* h3f = c2f + (size_t)BSZ * H2;
    float* c3f = h3f + (size_t)BSZ * H3;

    // ---- prep ----
    cvt_bf16_k<<<4096, 256, 0, stream>>>(x, xbf, (long)T * BSZ * IN);
    build_wcat_k<<<4096, 256, 0, stream>>>(Wih1, Whh1, w1, IN, H1, IN + H1, (long)4 * H1 * (IN + H1));
    build_wcat_k<<<4096, 256, 0, stream>>>(Wih2, Whh2, w2, H1, H2, H1 + H2, (long)4 * H2 * (H1 + H2));
    build_wcat_k<<<4096, 256, 0, stream>>>(Wih3, Whh3, w3, H2, H3, H2 + H3, (long)4 * H3 * (H2 + H3));
    bias_sum_k<<<(4 * H1 + 255) / 256, 256, 0, stream>>>(bih1, bhh1, b1, 4 * H1);
    bias_sum_k<<<(4 * H2 + 255) / 256, 256, 0, stream>>>(bih2, bhh2, b2, 4 * H2);
    bias_sum_k<<<(4 * H3 + 255) / 256, 256, 0, stream>>>(bih3, bhh3, b3, 4 * H3);
    zero_f32_k<<<1024, 256, 0, stream>>>((float*)d_out, (long)out_size);
    zero_bf16_k<<<1024, 256, 0, stream>>>(h1bf, (long)BSZ * (H1 + H2 + H3)); // contiguous

    const int K1 = IN + H1, N1 = 4 * H1;
    const int K2 = H1 + H2, N2 = 4 * H2;
    const int K3 = H2 + H3, N3 = 4 * H3;

    for (int t = 0; t < T; ++t) {
        const bf16* xt = xbf + (size_t)t * BSZ * IN;
        // layer 1
        lstm_gates_gemm<<<dim3(N1 / 64, 2), 256, 0, stream>>>(xt, IN, IN, h1bf, H1,
                                                              w1, K1, N1, gbuf);
        lstm_pointwise<<<(BSZ * H1) / 256, 256, 0, stream>>>(gbuf, b1, c1f, h1f, h1bf, H1);
        // layer 2
        lstm_gates_gemm<<<dim3(N2 / 64, 2), 256, 0, stream>>>(h1bf, H1, H1, h2bf, H2,
                                                              w2, K2, N2, gbuf);
        lstm_pointwise<<<(BSZ * H2) / 256, 256, 0, stream>>>(gbuf, b2, c2f, h2f, h2bf, H2);
        // layer 3
        lstm_gates_gemm<<<dim3(N3 / 64, 2), 256, 0, stream>>>(h2bf, H2, H2, h3bf, H3,
                                                              w3, K3, N3, gbuf);
        lstm_pointwise<<<(BSZ * H3) / 256, 256, 0, stream>>>(gbuf, b3, c3f, h3f, h3bf, H3);
    }
}
